// EncoderLayer_25890062860749
// MI455X (gfx1250) — matrix operand order repaired and checked by reading, not compiled, not run on hardware
//
#include <hip/hip_runtime.h>
#include <hip/hip_bf16.h>

// ---------------- problem constants ----------------
#define BB 8
#define TT_ 1024
#define FF 512
#define HH 8
#define DKK 64
#define EPSV 1e-5f

typedef unsigned short ushort_t;

typedef __attribute__((ext_vector_type(16))) __bf16 v16bf;
typedef __attribute__((ext_vector_type(8)))  float  v8f;

struct __attribute__((aligned(16))) U128 { unsigned int x0, x1, x2, x3; };
struct U256 { U128 lo, hi; };

__device__ __forceinline__ v16bf mkfrag(U128 lo, U128 hi) {
    U256 u; u.lo = lo; u.hi = hi;
    return __builtin_bit_cast(v16bf, u);
}

__device__ __forceinline__ ushort_t f2bf(float f) {
    unsigned int u = __float_as_uint(f);
    unsigned int r = (u + 0x7FFFu + ((u >> 16) & 1u)) >> 16;
    return (ushort_t)r;
}

// ---------------- conversion kernels ----------------
__global__ __launch_bounds__(256) void cvt_bf16_kernel(const float* __restrict__ x,
                                                       ushort_t* __restrict__ y, int n) {
    int i = blockIdx.x * 256 + threadIdx.x;
    if (i < n) y[i] = f2bf(x[i]);
}

// W is K x N row-major (fp32); output Wt is N x K row-major (bf16)
__global__ __launch_bounds__(256) void cvt_transpose_kernel(const float* __restrict__ W,
                                                            ushort_t* __restrict__ Wt,
                                                            int K, int N) {
    int i = blockIdx.x * 256 + threadIdx.x;
    if (i < K * N) {
        int k = i / N, n = i % N;
        Wt[(size_t)n * K + k] = f2bf(W[i]);
    }
}

// ---------------- WMMA bf16 GEMM ----------------
// C[M x N] (f32, +bias) = A[M x K] (bf16 row-major) * Bt[N x K]^T (bf16)
// block = 256 threads (8 waves); tile M=128, N=64, Kstep=32
#define MT 128
#define NT 64
#define KT 32

__global__ __launch_bounds__(256) void gemm_bf16_kernel(const ushort_t* __restrict__ A,
                                                        const ushort_t* __restrict__ Bt,
                                                        const float* __restrict__ bias,
                                                        float* __restrict__ C,
                                                        int M, int N, int K) {
    __shared__ __attribute__((aligned(16))) ushort_t Alds[MT * KT]; // 8 KB
    __shared__ __attribute__((aligned(16))) ushort_t Blds[NT * KT]; // 4 KB

    const int tid  = threadIdx.x;
    const int wave = tid >> 5;
    const int lane = tid & 31;
    const int l15  = lane & 15;
    const int hi   = lane >> 4;

    const int m0 = blockIdx.x * MT;
    const int n0 = blockIdx.y * NT;
    const int wm = (wave & 3) * 32;   // 4 waves tile M
    const int wn = (wave >> 2) * 32;  // 2 waves tile N

    v8f acc[2][2] = {};

    const int arow = tid >> 1, ahalf = (tid & 1) * 16;   // A: 128 rows x 2 halves
    const int brow = tid >> 1, bhalf = (tid & 1) * 16;   // B: 64 rows x 2 halves (tid<128)

    for (int k0 = 0; k0 < K; k0 += KT) {
        // stage A tile (128 x 32)
        {
            const U128* src = (const U128*)(A + (size_t)(m0 + arow) * K + k0 + ahalf);
            U128 a0 = src[0], a1 = src[1];
            *(U128*)(&Alds[arow * KT + ahalf])     = a0;
            *(U128*)(&Alds[arow * KT + ahalf + 8]) = a1;
            if (k0 + KT < K)
                __builtin_prefetch((const void*)(A + (size_t)(m0 + arow) * K + k0 + KT + ahalf), 0, 1);
        }
        // stage B tile (64 x 32) from pre-transposed weights
        if (tid < 128) {
            const U128* src = (const U128*)(Bt + (size_t)(n0 + brow) * K + k0 + bhalf);
            U128 b0 = src[0], b1 = src[1];
            *(U128*)(&Blds[brow * KT + bhalf])     = b0;
            *(U128*)(&Blds[brow * KT + bhalf + 8]) = b1;
        }
        __syncthreads();

        // B fragments: lanes 0-15 -> K {0..7,16..23} of column n; lanes 16-31 -> K {8..15,24..31}
        v16bf bfrag[2];
#pragma unroll
        for (int nj = 0; nj < 2; ++nj) {
            const ushort_t* bp = &Blds[(wn + nj * 16 + l15) * KT + hi * 8];
            bfrag[nj] = mkfrag(*(const U128*)bp, *(const U128*)(bp + 16));
        }
#pragma unroll
        for (int mi = 0; mi < 2; ++mi) {
            // A fragment: lanes 0-15 -> K {0..7,16..23}; lanes 16-31 -> K {8..15,24..31}
            const ushort_t* ap = &Alds[(wm + mi * 16 + l15) * KT];
            v16bf afrag = mkfrag(*(const U128*)(ap + hi * 8),
                                 *(const U128*)(ap + 16 + hi * 8));
#pragma unroll
            for (int nj = 0; nj < 2; ++nj) {
                acc[mi][nj] = __builtin_amdgcn_wmma_f32_16x16x32_bf16(
                    false, afrag, false, bfrag[nj], (short)0, acc[mi][nj], false, false);
            }
        }
        __syncthreads();
    }

    // epilogue: D layout -> lane n = lane&15, rows (hi?8..15:0..7)
#pragma unroll
    for (int mi = 0; mi < 2; ++mi) {
#pragma unroll
        for (int nj = 0; nj < 2; ++nj) {
            int n = n0 + wn + nj * 16 + l15;
            float bv = bias ? bias[n] : 0.0f;
            int mtop = m0 + wm + mi * 16 + hi * 8;
#pragma unroll
            for (int r = 0; r < 8; ++r)
                C[(size_t)(mtop + r) * N + n] = acc[mi][nj][r] + bv;
        }
    }
}

// ---------------- column mean / rstd over rows (BN training stats) ----------------
__global__ __launch_bounds__(256) void colstats_kernel(const float* __restrict__ X,
                                                       int R, int Ccols,
                                                       float* __restrict__ mean,
                                                       float* __restrict__ rstd) {
    int c = blockIdx.x;
    int tid = threadIdx.x;
    float s1 = 0.f, s2 = 0.f;
    for (int r = tid; r < R; r += 256) {
        float v = X[(size_t)r * Ccols + c];
        s1 += v; s2 += v * v;
    }
    __shared__ float sh1[256], sh2[256];
    sh1[tid] = s1; sh2[tid] = s2;
    __syncthreads();
    for (int s = 128; s > 0; s >>= 1) {
        if (tid < s) { sh1[tid] += sh1[tid + s]; sh2[tid] += sh2[tid + s]; }
        __syncthreads();
    }
    if (tid == 0) {
        float mu  = sh1[0] / (float)R;
        float var = sh2[0] / (float)R - mu * mu;   // biased, matches jnp.var
        mean[c] = mu;
        rstd[c] = rsqrtf(var + EPSV);
    }
}

// ---------------- softmax over t for k, with raw-reshape gather ----------------
// klin: (B*T) x 64 ; karr: (B, DK, T). k[b,kk,t] = klin[b*1024 + 16*kk + (t>>6)][t&63]
__global__ __launch_bounds__(256) void softmax_k_kernel(const float* __restrict__ klin,
                                                        float* __restrict__ karr) {
    int b  = blockIdx.x >> 6;
    int kk = blockIdx.x & 63;
    int tid = threadIdx.x;
    __shared__ float zbuf[1024];
    __shared__ float red[256];

    float lmax = -3.4e38f;
    for (int t = tid; t < 1024; t += 256) {
        int row = b * 1024 + kk * 16 + (t >> 6);
        int col = t & 63;
        float v = klin[(size_t)row * 64 + col];
        zbuf[t] = v;
        lmax = fmaxf(lmax, v);
    }
    red[tid] = lmax;
    __syncthreads();
    for (int s = 128; s > 0; s >>= 1) {
        if (tid < s) red[tid] = fmaxf(red[tid], red[tid + s]);
        __syncthreads();
    }
    float m = red[0];
    __syncthreads();

    float lsum = 0.f;
    for (int t = tid; t < 1024; t += 256) {
        float e = __expf(zbuf[t] - m);
        zbuf[t] = e;
        lsum += e;
    }
    red[tid] = lsum;
    __syncthreads();
    for (int s = 128; s > 0; s >>= 1) {
        if (tid < s) red[tid] += red[tid + s];
        __syncthreads();
    }
    float inv = 1.0f / red[0];
    float* out = karr + ((size_t)b * 64 + kk) * 1024;
    for (int t = tid; t < 1024; t += 256) out[t] = zbuf[t] * inv;
}

// ---------------- BN(v) + rearrange to (B, DK, T) ----------------
__global__ __launch_bounds__(256) void vbn_kernel(const float* __restrict__ vlin,
                                                  const float* __restrict__ vmean,
                                                  const float* __restrict__ vrstd,
                                                  const float* __restrict__ gv,
                                                  const float* __restrict__ betav,
                                                  float* __restrict__ varr) {
    int i = blockIdx.x * 256 + threadIdx.x;   // i over B*DK*T = 524288
    if (i >= BB * DKK * TT_) return;
    int b  = i >> 16;
    int kk = (i >> 10) & 63;
    int t  = i & 1023;
    int row = b * 1024 + kk * 16 + (t >> 6);
    int col = t & 63;
    float v = vlin[(size_t)row * 64 + col];
    varr[i] = (v - vmean[col]) * vrstd[col] * gv[col] + betav[col];
}

// ---------------- lam_c[b,k,v] = sum_t karr[b,k,t] * varr[b,v,t] ----------------
__global__ __launch_bounds__(256) void lamc_kernel(const float* __restrict__ karr,
                                                   const float* __restrict__ varr,
                                                   float* __restrict__ lamc) {
    int b = blockIdx.x >> 6;
    int k = blockIdx.x & 63;
    int tid = threadIdx.x;
    __shared__ float ksh[1024];
    for (int t = tid; t < 1024; t += 256)
        ksh[t] = karr[((size_t)b * 64 + k) * 1024 + t];
    __syncthreads();

    int wave = tid >> 5, lane = tid & 31;
    for (int v = wave; v < 64; v += 8) {
        const float* vp = varr + ((size_t)b * 64 + v) * 1024;
        float s = 0.f;
        for (int t = lane; t < 1024; t += 32) s += ksh[t] * vp[t];
#pragma unroll
        for (int off = 16; off > 0; off >>= 1) s += __shfl_down(s, off, 32);
        if (lane == 0) lamc[(((size_t)b * 64) + k) * 64 + v] = s;
    }
}

// ---------------- fused context: conv3x3+relu+lam_c, contract with BN(q), emit bf16 Y^T ----------------
// out Ytb: (B, T, F) bf16, f = h*64 + v
#define CTT 16
__global__ __launch_bounds__(256) void context_kernel(const float* __restrict__ qlin,
                                                      const float* __restrict__ qmean,
                                                      const float* __restrict__ qrstd,
                                                      const float* __restrict__ gq,
                                                      const float* __restrict__ betaq,
                                                      const float* __restrict__ varr,
                                                      const float* __restrict__ lamc,
                                                      const float* __restrict__ Wpos,
                                                      ushort_t* __restrict__ Ytb) {
    int b  = blockIdx.y;
    int t0 = blockIdx.x * CTT;
    int tid = threadIdx.x;

    __shared__ float qs[CTT][FF];      // 32 KB, BN-applied q for 16 timesteps
    __shared__ float lcs[DKK * DKK];   // 16 KB
    __shared__ float vimg[66][18];     // padded v image slice
    __shared__ float wps[DKK * 9];     // conv weights

    for (int i = tid; i < DKK * DKK; i += 256) lcs[i] = lamc[(size_t)b * DKK * DKK + i];
    for (int i = tid; i < DKK * 9;  i += 256) wps[i] = Wpos[i];
    for (int i = tid; i < 66 * 18;  i += 256) {
        int rr = i / 18 - 1;
        int cc = i % 18 - 1 + t0;
        float v = 0.f;
        if (rr >= 0 && rr < 64 && cc >= 0 && cc < 1024)
            v = varr[((size_t)b * 64 + rr) * 1024 + cc];
        vimg[i / 18][i % 18] = v;
    }
    // q gather with raw-reshape index + BN:
    // q[b,h,o,t] = qlin[b*1024 + h*128 + 2*o + (t>>9)][t&511]
    for (int i = tid; i < CTT * FF; i += 256) {
        int tt = i >> 9;
        int ho = i & 511;
        int t  = t0 + tt;
        int h = ho >> 6, o = ho & 63;
        int row = b * 1024 + h * 128 + o * 2 + (t >> 9);
        int col = t & 511;
        float qv = qlin[(size_t)row * FF + col];
        qs[tt][ho] = (qv - qmean[col]) * qrstd[col] * gq[col] + betaq[col];
    }
    __syncthreads();

    for (int item = tid; item < CTT * 64; item += 256) {
        int tt = item >> 6;
        int v  = item & 63;
        float n00 = vimg[v + 0][tt + 0], n01 = vimg[v + 0][tt + 1], n02 = vimg[v + 0][tt + 2];
        float n10 = vimg[v + 1][tt + 0], n11 = vimg[v + 1][tt + 1], n12 = vimg[v + 1][tt + 2];
        float n20 = vimg[v + 2][tt + 0], n21 = vimg[v + 2][tt + 1], n22 = vimg[v + 2][tt + 2];

        float acc[HH];
#pragma unroll
        for (int h = 0; h < HH; ++h) acc[h] = 0.f;

        for (int o = 0; o < 64; ++o) {
            const float* w = &wps[o * 9];
            float cv = w[0] * n00 + w[1] * n01 + w[2] * n02
                     + w[3] * n10 + w[4] * n11 + w[5] * n12
                     + w[6] * n20 + w[7] * n21 + w[8] * n22;
            float m = lcs[o * 64 + v] + fmaxf(cv, 0.f);
#pragma unroll
            for (int h = 0; h < HH; ++h) acc[h] += qs[tt][h * 64 + o] * m;
        }
        size_t base = ((size_t)b * 1024 + (t0 + tt)) * FF;
#pragma unroll
        for (int h = 0; h < HH; ++h) Ytb[base + h * 64 + v] = f2bf(acc[h]);
    }
}

// ---------------- host-side orchestration ----------------
extern "C" void kernel_launch(void* const* d_in, const int* in_sizes, int n_in,
                              void* d_out, int out_size, void* d_ws, size_t ws_size,
                              hipStream_t stream) {
    const float* x     = (const float*)d_in[0];
    const float* Wq    = (const float*)d_in[1];
    const float* bq    = (const float*)d_in[2];
    const float* Wk    = (const float*)d_in[3];
    const float* bk    = (const float*)d_in[4];
    const float* Wv    = (const float*)d_in[5];
    const float* bv    = (const float*)d_in[6];
    const float* gq    = (const float*)d_in[7];
    const float* betaq = (const float*)d_in[8];
    const float* gv    = (const float*)d_in[9];
    const float* betav = (const float*)d_in[10];
    const float* Wpos  = (const float*)d_in[11];
    const float* Wout  = (const float*)d_in[12];
    const float* bout  = (const float*)d_in[13];
    float* out = (float*)d_out;

    const int M = BB * TT_;   // 8192
    const int K = FF;         // 512
    const int Nq = FF;        // 512
    const int Nkv = DKK;      // 64

    // carve workspace (256B aligned chunks)
    size_t off = 0;
    auto alloc = [&](size_t bytes) {
        void* p = (char*)d_ws + off;
        off += (bytes + 255) & ~(size_t)255;
        return p;
    };
    ushort_t* xb    = (ushort_t*)alloc((size_t)M * K * 2);
    ushort_t* WqT   = (ushort_t*)alloc((size_t)Nq * K * 2);
    ushort_t* WkT   = (ushort_t*)alloc((size_t)Nkv * K * 2);
    ushort_t* WvT   = (ushort_t*)alloc((size_t)Nkv * K * 2);
    ushort_t* WoutT = (ushort_t*)alloc((size_t)Nq * K * 2);
    float* qlin  = (float*)alloc((size_t)M * Nq * 4);
    float* klin  = (float*)alloc((size_t)M * Nkv * 4);
    float* vlin  = (float*)alloc((size_t)M * Nkv * 4);
    float* qmean = (float*)alloc(FF * 4);
    float* qrstd = (float*)alloc(FF * 4);
    float* vmean = (float*)alloc(DKK * 4);
    float* vrstd = (float*)alloc(DKK * 4);
    float* karr  = (float*)alloc((size_t)BB * DKK * TT_ * 4);
    float* varr  = (float*)alloc((size_t)BB * DKK * TT_ * 4);
    float* lamc  = (float*)alloc((size_t)BB * DKK * DKK * 4);
    ushort_t* Ytb = (ushort_t*)alloc((size_t)M * FF * 2);
    (void)ws_size; (void)n_in; (void)in_sizes; (void)out_size;

    // 1) convert inputs
    {
        int n = M * K;
        cvt_bf16_kernel<<<(n + 255) / 256, 256, 0, stream>>>(x, xb, n);
    }
    cvt_transpose_kernel<<<(K * Nq  + 255) / 256, 256, 0, stream>>>(Wq,   WqT,   K, Nq);
    cvt_transpose_kernel<<<(K * Nkv + 255) / 256, 256, 0, stream>>>(Wk,   WkT,   K, Nkv);
    cvt_transpose_kernel<<<(K * Nkv + 255) / 256, 256, 0, stream>>>(Wv,   WvT,   K, Nkv);
    cvt_transpose_kernel<<<(K * Nq  + 255) / 256, 256, 0, stream>>>(Wout, WoutT, K, Nq);

    // 2) q/k/v projections (WMMA bf16)
    gemm_bf16_kernel<<<dim3(M / MT, Nq  / NT), 256, 0, stream>>>(xb, WqT, bq, qlin, M, Nq,  K);
    gemm_bf16_kernel<<<dim3(M / MT, Nkv / NT), 256, 0, stream>>>(xb, WkT, bk, klin, M, Nkv, K);
    gemm_bf16_kernel<<<dim3(M / MT, Nkv / NT), 256, 0, stream>>>(xb, WvT, bv, vlin, M, Nkv, K);

    // 3) BN stats
    colstats_kernel<<<FF,  256, 0, stream>>>(qlin, M, FF,  qmean, qrstd);
    colstats_kernel<<<DKK, 256, 0, stream>>>(vlin, M, DKK, vmean, vrstd);

    // 4) softmax(k) and BN(v), rearranged to (B, DK, T)
    softmax_k_kernel<<<BB * DKK, 256, 0, stream>>>(klin, karr);
    {
        int n = BB * DKK * TT_;
        vbn_kernel<<<(n + 255) / 256, 256, 0, stream>>>(vlin, vmean, vrstd, gv, betav, varr);
    }

    // 5) lam_c
    lamc_kernel<<<BB * DKK, 256, 0, stream>>>(karr, varr, lamc);

    // 6) fused conv + context contraction -> bf16 Y^T (B,T,F)
    context_kernel<<<dim3(TT_ / CTT, BB), 256, 0, stream>>>(
        qlin, qmean, qrstd, gq, betaq, varr, lamc, Wpos, Ytb);

    // 7) output projection (WMMA bf16) + bias -> d_out (fp32)
    gemm_bf16_kernel<<<dim3(M / MT, Nq / NT), 256, 0, stream>>>(Ytb, WoutT, bout, out, M, Nq, K);
}
